// MoE_18872086298876
// MI455X (gfx1250) — matrix operand order repaired and checked by reading, not compiled, not run on hardware
//
#include <hip/hip_runtime.h>
#include <math.h>

// Problem constants (from reference): N=8192 tokens, D=1024, H=4096, E=8, k=2
#define N_TOK  8192
#define DMODEL 1024
#define DFF    4096
#define NEXP   8
#define MT     64      // token tile per workgroup
#define NWAVES 16      // 512 threads
#define HCHUNK 64      // d_ff chunk held in LDS between the two GEMMs

typedef _Float16 f16;
typedef __attribute__((ext_vector_type(16))) _Float16 v16h;
typedef __attribute__((ext_vector_type(8)))  _Float16 v8h;
typedef __attribute__((ext_vector_type(8)))  float    v8f;

__device__ __forceinline__ v8f zero8() {
  v8f z = {0.f, 0.f, 0.f, 0.f, 0.f, 0.f, 0.f, 0.f};
  return z;
}

// A-fragment (16x32 f16, M=lane&15) from row-major LDS, leading dim ld.
// Contiguous K order: lanes 0-15 hold K={0..15}, lanes 16-31 K={16..31};
// each 8-group is contiguous in LDS -> compiler merges into 2x ds_load_b128.
__device__ __forceinline__ v16h load_a_lds(const f16* __restrict__ A, int ld,
                                           int kbase, int lane) {
  const int half = lane >> 4;
  const f16* row = A + (size_t)(lane & 15) * ld;
  v16h r;
#pragma unroll
  for (int j = 0; j < 8; ++j) {
    const int k = kbase + (half << 4) + (j << 1);
    r[2 * j]     = row[k];
    r[2 * j + 1] = row[k + 1];
  }
  return r;
}

// B-fragment from pre-swizzled f16 weights: element (k,n) lives at
// ((k>>4)*ncols + n)*16 + (k&15). A lane's 16 values (k=half*16..+15, col n)
// are 32 contiguous bytes -> 2x global_load_b128, no VALU.
__device__ __forceinline__ v16h load_b_swz(const f16* __restrict__ B, int ncols,
                                           int kbase, int nbase, int lane) {
  const size_t k16 = (size_t)((kbase >> 4) + (lane >> 4));
  const size_t n = (size_t)(nbase + (lane & 15));
  return *(const v16h*)(B + ((k16 * ncols + n) << 4));
}

// Fallback: B-fragment straight from row-major fp32 weights (strided + cvt).
__device__ __forceinline__ v16h load_b_glb(const float* __restrict__ B, int ld,
                                           int kbase, int nbase, int lane) {
  const int half = lane >> 4;
  const float* p = B + (size_t)(kbase + (half << 4)) * ld + nbase + (lane & 15);
  v16h r;
#pragma unroll
  for (int j = 0; j < 8; ++j) {
    r[2 * j]     = (f16)p[(size_t)(2 * j) * ld];
    r[2 * j + 1] = (f16)p[(size_t)(2 * j + 1) * ld];
  }
  return r;
}

// ---------------------------------------------------------------------------
// Kernel 0: zero output accumulator and per-expert counters.
// ---------------------------------------------------------------------------
__global__ void moe_zero_kernel(float* __restrict__ out, int* __restrict__ cnt,
                                size_t n) {
  size_t i = (size_t)blockIdx.x * blockDim.x + threadIdx.x;
  if (i < n) out[i] = 0.f;
  if (i < NEXP) cnt[i] = 0;
}

// ---------------------------------------------------------------------------
// Kernel W: convert fp32 weights -> f16 in B-fragment-swizzled layout.
// One thread per (k16, n): 16 coalesced strided reads, one 32B store.
// ---------------------------------------------------------------------------
__global__ __launch_bounds__(256) void moe_cvt_swz_kernel(
    const float* __restrict__ src, f16* __restrict__ dst, int ncols,
    size_t total /* = nrows16 * ncols */) {
  size_t idx = (size_t)blockIdx.x * blockDim.x + threadIdx.x;
  if (idx >= total) return;
  const size_t n = idx % (size_t)ncols;
  const size_t k16 = idx / (size_t)ncols;
  const float* s = src + k16 * 16 * (size_t)ncols + n;
  v16h v;
#pragma unroll
  for (int i = 0; i < 16; ++i) v[i] = (f16)s[(size_t)i * ncols];
  *(v16h*)(dst + (idx << 4)) = v;
}

// Kernel X: plain fp32 -> f16 copy of activations.
__global__ void moe_cvt_x_kernel(const float* __restrict__ src,
                                 f16* __restrict__ dst, size_t n) {
  size_t i = (size_t)blockIdx.x * blockDim.x + threadIdx.x;
  if (i < n) dst[i] = (f16)src[i];
}

// ---------------------------------------------------------------------------
// Kernel 1: gating. One wave32 per token: logits (E=8), ||x||, top-2 softmax,
// append (token, gate) to per-expert lists with atomic counters.
// ---------------------------------------------------------------------------
__global__ __launch_bounds__(256) void moe_gate_kernel(
    const float* __restrict__ x, const float* __restrict__ wg,
    int* __restrict__ cnt, int* __restrict__ tok, float* __restrict__ gatew,
    float* __restrict__ xnorm) {
  const int wave = blockIdx.x * (blockDim.x >> 5) + (threadIdx.x >> 5);
  const int lane = threadIdx.x & 31;
  if (wave >= N_TOK) return;

  const float* xr = x + (size_t)wave * DMODEL;
  float acc[NEXP];
#pragma unroll
  for (int e = 0; e < NEXP; ++e) acc[e] = 0.f;
  float nrm = 0.f;

  for (int j = lane; j < DMODEL; j += 32) {
    const float xv = xr[j];
    nrm += xv * xv;
    const float* wr = wg + (size_t)j * NEXP;
#pragma unroll
    for (int e = 0; e < NEXP; ++e) acc[e] += xv * wr[e];
  }
#pragma unroll
  for (int off = 16; off >= 1; off >>= 1) {
    nrm += __shfl_xor(nrm, off, 32);
#pragma unroll
    for (int e = 0; e < NEXP; ++e) acc[e] += __shfl_xor(acc[e], off, 32);
  }

  if (lane == 0) {
    int e0 = 0;
    float v0 = acc[0];
#pragma unroll
    for (int e = 1; e < NEXP; ++e)
      if (acc[e] > v0) { v0 = acc[e]; e0 = e; }
    int e1 = (e0 == 0) ? 1 : 0;
    float v1 = acc[e1];
#pragma unroll
    for (int e = 0; e < NEXP; ++e)
      if (e != e0 && acc[e] > v1) { v1 = acc[e]; e1 = e; }
    const float a1 = __expf(v1 - v0);
    const float inv = 1.f / (1.f + a1);

    xnorm[wave] = sqrtf(nrm);
    int p0 = atomicAdd(&cnt[e0], 1);
    tok[e0 * N_TOK + p0] = wave;
    gatew[e0 * N_TOK + p0] = inv;
    int p1 = atomicAdd(&cnt[e1], 1);
    tok[e1 * N_TOK + p1] = wave;
    gatew[e1 * N_TOK + p1] = a1 * inv;
  }
}

// ---------------------------------------------------------------------------
// Kernel 2: per-expert fused 2-layer MLP on a 64-token tile via f16 WMMA.
// 16 waves / block. GEMM1: each wave owns one 16x16 tile of the 64x64 h-chunk.
// GEMM2: each wave owns 64 output columns (4 M x 4 N v8f accumulators).
// ---------------------------------------------------------------------------
template <bool SWZ>
__global__ __launch_bounds__(512) void moe_expert_kernel(
    const float* __restrict__ x, const f16* __restrict__ xh,
    const float* __restrict__ w1, const f16* __restrict__ w1h,
    const float* __restrict__ b1, const float* __restrict__ w2,
    const f16* __restrict__ w2h, const float* __restrict__ b2,
    const int* __restrict__ cnt, const int* __restrict__ tok,
    const float* __restrict__ gatew, const float* __restrict__ xnorm,
    float* __restrict__ out) {
  extern __shared__ char smem[];
  f16*   Xt    = (f16*)smem;                                    // MT x DMODEL (128KB)
  f16*   Hc    = (f16*)(smem + MT * DMODEL * 2);                // MT x HCHUNK (8KB)
  float* norm2 = (float*)(smem + MT * DMODEL * 2 + MT * HCHUNK * 2);
  float* scl   = norm2 + MT;
  int*   toks  = (int*)(scl + MT);
  float* gts   = (float*)(toks + MT);

  const int e = blockIdx.y;
  const int cnt_e = cnt[e];
  const int row0 = blockIdx.x * MT;
  if (row0 >= cnt_e) return;  // uniform early-exit

  const int tid = threadIdx.x;
  if (tid < MT) {
    const int r = row0 + tid;
    const bool v = r < cnt_e;
    toks[tid]  = v ? tok[e * N_TOK + r] : 0;
    gts[tid]   = v ? gatew[e * N_TOK + r] : 0.f;  // gate 0 kills padded rows
    norm2[tid] = 0.f;
  }
  __syncthreads();

  // Gather the 64 token rows into LDS (row-major f16).
  if (SWZ) {
    for (int i = tid; i < MT * DMODEL / 8; i += 512) {
      const int r = i >> 7, c8 = i & 127;  // DMODEL/8 = 128 chunks per row
      ((v8h*)Xt)[i] = *(const v8h*)(xh + (size_t)toks[r] * DMODEL + c8 * 8);
    }
  } else {
    for (int i = tid; i < MT * DMODEL; i += 512) {
      const int r = i >> 10, c = i & (DMODEL - 1);
      Xt[i] = (f16)x[(size_t)toks[r] * DMODEL + c];
    }
  }
  __syncthreads();

  const int wv = tid >> 5, lane = tid & 31;
  const int mt1 = wv >> 2, nt1 = wv & 3;  // GEMM1 (M-tile, N-tile) assignment
  const int half = lane >> 4, nn = lane & 15;
  const int colbase = wv * 64;            // GEMM2 output column range

  const float* w1e = w1 + (size_t)e * DMODEL * DFF;
  const float* w2e = w2 + (size_t)e * DFF * DMODEL;
  const f16* w1he = SWZ ? (w1h + (size_t)e * DMODEL * DFF) : nullptr;
  const f16* w2he = SWZ ? (w2h + (size_t)e * DFF * DMODEL) : nullptr;
  const float* b1e = b1 + (size_t)e * DFF;
  const float* b2e = b2 + (size_t)e * DMODEL;

  v8f acc[4][4];
#pragma unroll
  for (int m = 0; m < 4; ++m)
#pragma unroll
    for (int nt = 0; nt < 4; ++nt) acc[m][nt] = zero8();

  for (int hc = 0; hc < DFF; hc += HCHUNK) {
    // ---- GEMM1: h_chunk = relu(Xt @ w1[:, hc:hc+64] + b1) ----
    v8f ha = zero8();
    for (int ks = 0; ks < DMODEL; ks += 32) {
      v16h a = load_a_lds(Xt + mt1 * 16 * DMODEL, DMODEL, ks, lane);
      v16h b = SWZ ? load_b_swz(w1he, DFF, ks, hc + nt1 * 16, lane)
                   : load_b_glb(w1e, DFF, ks, hc + nt1 * 16, lane);
      ha = __builtin_amdgcn_wmma_f32_16x16x32_f16(false, a, false, b, (short)0,
                                                  ha, false, false);
    }
    __syncthreads();  // previous chunk's GEMM2 finished reading Hc
    {
      const float bb = b1e[hc + nt1 * 16 + nn];
#pragma unroll
      for (int r = 0; r < 8; ++r) {
        const float v = ha[r] + bb;
        Hc[(mt1 * 16 + half * 8 + r) * HCHUNK + nt1 * 16 + nn] =
            (f16)(v > 0.f ? v : 0.f);
      }
    }
    __syncthreads();

    // ---- GEMM2: acc += h_chunk @ w2[hc:hc+64, colbase:colbase+64] ----
#pragma unroll
    for (int ks2 = 0; ks2 < 2; ++ks2) {
      v16h a[4];
#pragma unroll
      for (int m = 0; m < 4; ++m)
        a[m] = load_a_lds(Hc + m * 16 * HCHUNK, HCHUNK, ks2 * 32, lane);
#pragma unroll
      for (int nt = 0; nt < 4; ++nt) {
        v16h b = SWZ ? load_b_swz(w2he, DMODEL, hc + ks2 * 32,
                                  colbase + nt * 16, lane)
                     : load_b_glb(w2e, DMODEL, hc + ks2 * 32,
                                  colbase + nt * 16, lane);
#pragma unroll
        for (int m = 0; m < 4; ++m)
          acc[m][nt] = __builtin_amdgcn_wmma_f32_16x16x32_f16(
              false, a[m], false, b, (short)0, acc[m][nt], false, false);
      }
    }
  }

  // ---- epilogue: + b2, row norms, renormalize to gate*||x||, scatter ----
#pragma unroll
  for (int nt = 0; nt < 4; ++nt) {
    const float bb = b2e[colbase + nt * 16 + nn];
#pragma unroll
    for (int m = 0; m < 4; ++m)
#pragma unroll
      for (int r = 0; r < 8; ++r) acc[m][nt][r] += bb;
  }
#pragma unroll
  for (int m = 0; m < 4; ++m) {
#pragma unroll
    for (int r = 0; r < 8; ++r) {
      float s = 0.f;
#pragma unroll
      for (int nt = 0; nt < 4; ++nt) s += acc[m][nt][r] * acc[m][nt][r];
#pragma unroll
      for (int off = 8; off >= 1; off >>= 1)  // reduce over the 16-lane half
        s += __shfl_xor(s, off, 32);
      if (nn == 0) atomicAdd(&norm2[m * 16 + half * 8 + r], s);
    }
  }
  __syncthreads();
  if (tid < MT) {
    const float on = sqrtf(norm2[tid]);
    scl[tid] = gts[tid] * xnorm[toks[tid]] / (on + 1e-8f);
  }
  __syncthreads();

#pragma unroll
  for (int m = 0; m < 4; ++m) {
#pragma unroll
    for (int r = 0; r < 8; ++r) {
      const int row = m * 16 + half * 8 + r;
      float* o = out + (size_t)toks[row] * DMODEL;
      const float sc = scl[row];
#pragma unroll
      for (int nt = 0; nt < 4; ++nt)
        atomicAdd(&o[colbase + nt * 16 + nn], acc[m][nt][r] * sc);
    }
  }
}

// ---------------------------------------------------------------------------
extern "C" void kernel_launch(void* const* d_in, const int* in_sizes, int n_in,
                              void* d_out, int out_size, void* d_ws,
                              size_t ws_size, hipStream_t stream) {
  const float* x  = (const float*)d_in[0];
  const float* wg = (const float*)d_in[1];
  const float* w1 = (const float*)d_in[2];
  const float* b1 = (const float*)d_in[3];
  const float* w2 = (const float*)d_in[4];
  const float* b2 = (const float*)d_in[5];
  // d_in[6] is k; fixed at 2 for this problem.
  float* out = (float*)d_out;

  char* ws = (char*)d_ws;
  int*   cnt   = (int*)ws;
  int*   tok   = (int*)(ws + 256);
  float* gatew = (float*)(ws + 256 + sizeof(int) * NEXP * N_TOK);
  float* xnorm = (float*)(ws + 256 + 2 * sizeof(int) * NEXP * N_TOK);
  size_t off = 256 + (size_t)2 * sizeof(int) * NEXP * N_TOK + 4 * N_TOK; // 557,312 (256B aligned)
  f16* xh  = (f16*)(ws + off);
  f16* w1h = (f16*)(ws + off + (size_t)2 * N_TOK * DMODEL);
  f16* w2h = (f16*)(ws + off + (size_t)2 * N_TOK * DMODEL +
                    (size_t)2 * NEXP * DMODEL * DFF);
  const size_t need = off + (size_t)2 * N_TOK * DMODEL +
                      (size_t)4 * NEXP * DMODEL * DFF;  // ~145 MB
  const bool use_f16w = ws_size >= need;

  const size_t nout = (size_t)N_TOK * DMODEL;
  moe_zero_kernel<<<(unsigned)((nout + 255) / 256), 256, 0, stream>>>(out, cnt,
                                                                      nout);
  moe_gate_kernel<<<N_TOK / 8, 256, 0, stream>>>(x, wg, cnt, tok, gatew, xnorm);

  const size_t smem =
      (size_t)MT * DMODEL * 2 + MT * HCHUNK * 2 + MT * 16 + 64;  // ~140 KB
  dim3 grid(N_TOK / MT, NEXP);

  if (use_f16w) {
    const size_t nx = (size_t)N_TOK * DMODEL;
    moe_cvt_x_kernel<<<(unsigned)((nx + 255) / 256), 256, 0, stream>>>(x, xh, nx);
    const size_t t1 = (size_t)(NEXP * DMODEL / 16) * DFF;  // (k16,n) pairs
    moe_cvt_swz_kernel<<<(unsigned)((t1 + 255) / 256), 256, 0, stream>>>(
        w1, w1h, DFF, t1);
    const size_t t2 = (size_t)(NEXP * DFF / 16) * DMODEL;
    moe_cvt_swz_kernel<<<(unsigned)((t2 + 255) / 256), 256, 0, stream>>>(
        w2, w2h, DMODEL, t2);
    moe_expert_kernel<true><<<grid, 512, smem, stream>>>(
        x, xh, w1, w1h, b1, w2, w2h, b2, cnt, tok, gatew, xnorm, out);
  } else {
    moe_expert_kernel<false><<<grid, 512, smem, stream>>>(
        x, nullptr, w1, nullptr, b1, w2, nullptr, b2, cnt, tok, gatew, xnorm,
        out);
  }
}
